// ChunkLevelFeatureEncoderAttentionV3_14310831030950
// MI455X (gfx1250) — matrix operand order repaired and checked by reading, not compiled, not run on hardware
//
#include <hip/hip_runtime.h>
#include <hip/hip_bf16.h>

// ---------------------------------------------------------------------------
// MI455X (gfx1250) implementation.
// - All heavy GEMMs: v_wmma_f32_16x16x32_bf16 (bf16 A/B, f32 accum)
// - GEMM tile staging: TDM tensor_load_to_lds (TENSORcnt), double buffered,
//   with D# LDS padding used for bank-conflict-free fragment reads.
// - Layer-2 attention: fused flash kernel; V tile via async global->LDS.
// ---------------------------------------------------------------------------

typedef __bf16 v16bf __attribute__((ext_vector_type(16)));
typedef float  v8f   __attribute__((ext_vector_type(8)));
typedef unsigned int v4u __attribute__((ext_vector_type(4)));
typedef int v8i __attribute__((ext_vector_type(8)));
typedef int v4i __attribute__((ext_vector_type(4)));

union BF16Frag {
    v16bf v;
    uint4 u4[2];
    unsigned short s[16];
};

static __device__ __forceinline__ unsigned short f2b(float f) {
    unsigned int u = __float_as_uint(f);
    unsigned int r = (u + 0x7FFFu + ((u >> 16) & 1u)) >> 16;  // RNE
    return (unsigned short)r;
}
static __device__ __forceinline__ float b2f(unsigned short h) {
    return __uint_as_float(((unsigned int)h) << 16);
}

static __device__ __forceinline__ v8f wmma_bf16(const v16bf& a, const v16bf& b, const v8f& c) {
    return __builtin_amdgcn_wmma_f32_16x16x32_bf16(false, a, false, b, (short)0, c, false, false);
}

// ---------------------------------------------------------------------------
// TDM: issue a 2D tile load (bf16 elements) global -> LDS via Tensor Data
// Mover.  D# packing per CDNA5 ISA 8.3/8.4 (group0 128b, group1 256b).
// pad_interval_code / pad_amount_code per ISA tables (DWORD units).
// ---------------------------------------------------------------------------
static __device__ __forceinline__ void tdm_load_2d(unsigned lds_addr,
                                                   const unsigned short* gptr,
                                                   unsigned tensor_d0, unsigned tensor_d1,
                                                   unsigned tile_d0, unsigned tile_d1,
                                                   unsigned stride_d0,
                                                   unsigned pad_interval_code,
                                                   unsigned pad_amount_code) {
    unsigned long long ga = (unsigned long long)(size_t)gptr;      // byte address
    v4u g0;
    g0.x = 1u;                                                     // count=1, user D#
    g0.y = lds_addr;                                               // lds_addr [63:32]
    g0.z = (unsigned)ga;                                           // global_addr [95:64]
    g0.w = (unsigned)((ga >> 32) & 0x01FFFFFFull) | (2u << 30);    // addr hi + type=2
    unsigned w0 = (1u << 16)                                       // data_size: 2 bytes
                | (1u << 20)                                       // pad_enable
                | (pad_interval_code << 22)
                | (pad_amount_code << 25);
    unsigned w1 = (tensor_d0 & 0xFFFFu) << 16;                     // tensor_dim0 lo
    unsigned w2 = (tensor_d0 >> 16) | ((tensor_d1 & 0xFFFFu) << 16);
    unsigned w3 = (tensor_d1 >> 16) | (tile_d0 << 16);             // tile_dim0
    unsigned w4 = tile_d1;                                         // tile_dim1, tile_dim2=0
    unsigned w5 = stride_d0;                                       // tensor_dim0_stride lo
    v8i g1 = {(int)w0, (int)w1, (int)w2, (int)w3, (int)w4, (int)w5, 0, 0};
#if defined(__clang_major__) && __clang_major__ >= 23
    __builtin_amdgcn_tensor_load_to_lds(g0, g1, (v4i)0, (v4i)0, (v8i)0, 0);
#else
    __builtin_amdgcn_tensor_load_to_lds(g0, g1, (v4i)0, (v4i)0, 0);
#endif
}

#define MB 16
#define SB 512
#define HB 768
#define CB 64
#define LB 8
#define NHB 8
#define DHB 96
#define FFB 3072
#define MROWS 8192   // == B*C*L == B*S

// ---------------------------------------------------------------------------
// f32 -> bf16 convert
// ---------------------------------------------------------------------------
__global__ void cvt_kernel(const float* __restrict__ src, unsigned short* __restrict__ dst, long n) {
    long i = (long)blockIdx.x * 256 + threadIdx.x;
    long stride = (long)gridDim.x * 256;
    for (; i < n; i += stride) dst[i] = f2b(src[i]);
}

// ---------------------------------------------------------------------------
// Gather chunk tokens: feat[(b*C+c)*L + p][:] = valid ? tlf[b][off+p][:] : 0
// ---------------------------------------------------------------------------
__global__ __launch_bounds__(256) void gather_kernel(const float* __restrict__ tlf,
                                                     const int* __restrict__ cl,
                                                     float* __restrict__ xg,
                                                     unsigned short* __restrict__ xg16) {
    int bc = blockIdx.x;
    int b = bc >> 6, c = bc & 63;
    const int* clb = cl + b * CB;
    int off = 0;
    for (int i = 0; i < c; ++i) off += clb[i];       // exclusive cumsum (raw cl)
    int len = clb[c]; if (len > LB) len = LB;
    int t = threadIdx.x;
    for (int p = 0; p < LB; ++p) {
        int idx = off + p;
        bool valid = p < len;
        int srow = idx < 0 ? 0 : (idx > SB - 1 ? SB - 1 : idx);
        const float* src = tlf + ((size_t)b * SB + srow) * HB;
        size_t drow = ((size_t)bc * LB + p) * HB;
        for (int i = t; i < HB; i += 256) {
            float v = valid ? src[i] : 0.0f;
            xg[drow + i] = v;
            xg16[drow + i] = f2b(v);
        }
    }
}

// ---------------------------------------------------------------------------
// Scatter layer-1 output into span (xg/xg16 must be pre-zeroed)
// ---------------------------------------------------------------------------
__global__ __launch_bounds__(256) void scatter_kernel(const float* __restrict__ feat,
                                                      const int* __restrict__ cl,
                                                      float* __restrict__ xg,
                                                      unsigned short* __restrict__ xg16) {
    int bc = blockIdx.x;
    int b = bc >> 6, c = bc & 63;
    const int* clb = cl + b * CB;
    int off = 0;
    for (int i = 0; i < c; ++i) off += clb[i];
    int len = clb[c]; if (len > LB) len = LB;
    int t = threadIdx.x;
    for (int p = 0; p < len; ++p) {
        int idx = off + p;
        if (idx >= SB) continue;                     // drop OOB
        const float* src = feat + ((size_t)bc * LB + p) * HB;
        size_t drow = ((size_t)b * SB + idx) * HB;
        for (int i = t; i < HB; i += 256) {
            float v = src[i];
            xg[drow + i] = v;
            xg16[drow + i] = f2b(v);
        }
    }
}

// ---------------------------------------------------------------------------
// WMMA GEMM: C[M,N] = A[M,K](bf16) * B[K,N](bf16) + bias, optional exact GELU.
// 128x128 tile per 256-thread block; 8 waves, each 32x64 (2x4 WMMA accs).
// Tiles staged by TDM (wave 0 issues descriptors), double buffered;
// D# padding gives the bank-conflict pads in LDS.
// ---------------------------------------------------------------------------
#define LDA_P 40    // 32 + 8 bf16 pad  (pad = 16B = 4 DWORDs)
#define LDB_P 136   // 128 + 8 bf16 pad

__global__ __launch_bounds__(256) void gemm_bf16_kernel(const unsigned short* __restrict__ A,
                                                        const unsigned short* __restrict__ B,
                                                        const float* __restrict__ bias,
                                                        float* __restrict__ outF,
                                                        unsigned short* __restrict__ outB,
                                                        int M, int N, int K, int gelu) {
    __shared__ unsigned short lA[2][128 * LDA_P];
    __shared__ unsigned short lB[2][32 * LDB_P];
    int t = threadIdx.x;
    int w = t >> 5;
    int l = t & 31;
    int r = l & 15, hi = l >> 4;
    int m0 = blockIdx.x * 128, n0 = blockIdx.y * 128;
    int sub_m = (w & 3) * 32;
    int sub_n = (w >> 2) * 64;

    unsigned ldsA[2] = {(unsigned)(size_t)&lA[0][0], (unsigned)(size_t)&lA[1][0]};
    unsigned ldsB[2] = {(unsigned)(size_t)&lB[0][0], (unsigned)(size_t)&lB[1][0]};

    v8f acc[2][4];
    const v8f vzero = {0.f, 0.f, 0.f, 0.f, 0.f, 0.f, 0.f, 0.f};
    for (int mi = 0; mi < 2; ++mi)
        for (int ni = 0; ni < 4; ++ni) acc[mi][ni] = vzero;

    // Preload tile 0 via TDM.
    // A tile: 128 rows x 32 cols, row = 16 DWORDs -> interval code 3, pad 4 DW -> code 3
    // B tile: 32 rows x 128 cols, row = 64 DWORDs -> interval code 5, pad 4 DW -> code 3
    if (w == 0) {
        tdm_load_2d(ldsA[0], A + (size_t)m0 * K, (unsigned)K, (unsigned)M, 32u, 128u, (unsigned)K, 3u, 3u);
        tdm_load_2d(ldsB[0], B + (size_t)0 * N + n0, (unsigned)N, (unsigned)K, 128u, 32u, (unsigned)N, 5u, 3u);
    }

    int buf = 0;
    for (int k0 = 0; k0 < K; k0 += 32, buf ^= 1) {
        bool hasNext = (k0 + 32) < K;
        if (w == 0) {
            if (hasNext) {
                tdm_load_2d(ldsA[buf ^ 1], A + (size_t)m0 * K + (k0 + 32),
                            (unsigned)K, (unsigned)M, 32u, 128u, (unsigned)K, 3u, 3u);
                tdm_load_2d(ldsB[buf ^ 1], B + (size_t)(k0 + 32) * N + n0,
                            (unsigned)N, (unsigned)K, 128u, 32u, (unsigned)N, 5u, 3u);
                __builtin_amdgcn_s_wait_tensorcnt(2);   // current-buf tiles complete
            } else {
                __builtin_amdgcn_s_wait_tensorcnt(0);
            }
        }
        __syncthreads();

        const unsigned short* lAb = &lA[buf][0];
        const unsigned short* lBb = &lB[buf][0];
        BF16Frag af[2], bfB[4];
        for (int mi = 0; mi < 2; ++mi) {
            const unsigned short* p = &lAb[(sub_m + mi * 16 + r) * LDA_P + hi * 8];
            af[mi].u4[0] = *(const uint4*)p;          // K  0..7  / 8..15
            af[mi].u4[1] = *(const uint4*)(p + 16);   // K 16..23 / 24..31
        }
        for (int ni = 0; ni < 4; ++ni) {
            const unsigned short* p = &lBb[hi * 8 * LDB_P + sub_n + ni * 16 + r];
            for (int s = 0; s < 8; ++s) {
                bfB[ni].s[s] = p[s * LDB_P];              // K 0..7 / 8..15
                bfB[ni].s[8 + s] = p[(16 + s) * LDB_P];   // K 16..23 / 24..31
            }
        }
        for (int mi = 0; mi < 2; ++mi)
            for (int ni = 0; ni < 4; ++ni)
                acc[mi][ni] = wmma_bf16(af[mi].v, bfB[ni].v, acc[mi][ni]);
        __syncthreads();   // all waves done reading buf before TDM overwrites it
    }

    // epilogue: C/D layout -> row = rr + 8*hi, col = lane%16
    for (int mi = 0; mi < 2; ++mi) {
        for (int ni = 0; ni < 4; ++ni) {
            for (int rr = 0; rr < 8; ++rr) {
                int row = m0 + sub_m + mi * 16 + rr + 8 * hi;
                int col = n0 + sub_n + ni * 16 + r;
                float v = acc[mi][ni][rr] + bias[col];
                if (gelu) v = 0.5f * v * (1.0f + erff(v * 0.70710678118654752f));
                size_t o = (size_t)row * N + col;
                if (outF) outF[o] = v;
                if (outB) outB[o] = f2b(v);
            }
        }
    }
}

// ---------------------------------------------------------------------------
// Residual add + LayerNorm over H=768; one block per row.
// ---------------------------------------------------------------------------
__global__ __launch_bounds__(256) void ln_kernel(const float* __restrict__ y,
                                                 const float* __restrict__ res,
                                                 const float* __restrict__ g,
                                                 const float* __restrict__ b,
                                                 float* __restrict__ outF,
                                                 unsigned short* __restrict__ outB) {
    __shared__ float red[256];
    int row = blockIdx.x;
    int t = threadIdx.x;
    const float* yr = y + (size_t)row * HB;
    const float* rr = res ? res + (size_t)row * HB : nullptr;
    float vals[3];
    float s = 0.0f;
    for (int i = 0; i < 3; ++i) {
        int c = t + i * 256;
        float v = yr[c] + (rr ? rr[c] : 0.0f);
        vals[i] = v;
        s += v;
    }
    red[t] = s; __syncthreads();
    for (int off = 128; off > 0; off >>= 1) { if (t < off) red[t] += red[t + off]; __syncthreads(); }
    float mean = red[0] * (1.0f / HB);
    __syncthreads();
    float s2 = 0.0f;
    for (int i = 0; i < 3; ++i) { float d = vals[i] - mean; s2 += d * d; }
    red[t] = s2; __syncthreads();
    for (int off = 128; off > 0; off >>= 1) { if (t < off) red[t] += red[t + off]; __syncthreads(); }
    float inv = rsqrtf(red[0] * (1.0f / HB) + 1e-12f);
    for (int i = 0; i < 3; ++i) {
        int c = t + i * 256;
        float o = (vals[i] - mean) * inv * g[c] + b[c];
        size_t idx = (size_t)row * HB + c;
        if (outF) outF[idx] = o;
        if (outB) outB[idx] = f2b(o);
    }
}

// ---------------------------------------------------------------------------
// Layer-1 attention: per chunk (seq len 8) x 8 heads. Tiny -> scalar in LDS.
// Soft -10000 mask over invalid k positions (matches reference exactly).
// ---------------------------------------------------------------------------
__global__ __launch_bounds__(256) void chunk_attn_kernel(const unsigned short* __restrict__ Q,
                                                         const unsigned short* __restrict__ K,
                                                         const unsigned short* __restrict__ V,
                                                         const int* __restrict__ cl,
                                                         unsigned short* __restrict__ ctx16) {
    __shared__ float qs[LB * DHB], ks[LB * DHB], vs[LB * DHB];
    __shared__ float sc[LB * LB], ps[LB * LB];
    int bc = blockIdx.x;
    int b = bc >> 6, c = bc & 63;
    int len = cl[b * CB + c]; if (len > LB) len = LB;
    int t = threadIdx.x;
    const float scale = 0.1020620726159658f;  // 1/sqrt(96)
    size_t base = (size_t)bc * LB * HB;
    for (int h = 0; h < NHB; ++h) {
        int h0 = h * DHB;
        for (int e = t; e < LB * DHB; e += 256) {
            int rrow = e / DHB, d = e % DHB;
            size_t src = base + (size_t)rrow * HB + h0 + d;
            qs[e] = b2f(Q[src]); ks[e] = b2f(K[src]); vs[e] = b2f(V[src]);
        }
        __syncthreads();
        if (t < 64) {
            int qi = t >> 3, ki = t & 7;
            float s = 0.0f;
            for (int d = 0; d < DHB; ++d) s += qs[qi * DHB + d] * ks[ki * DHB + d];
            sc[t] = s * scale + (ki < len ? 0.0f : -10000.0f);
        }
        __syncthreads();
        if (t < 8) {
            float m = -1e30f;
            for (int k = 0; k < LB; ++k) m = fmaxf(m, sc[t * LB + k]);
            float e[LB], sum = 0.0f;
            for (int k = 0; k < LB; ++k) { e[k] = expf(sc[t * LB + k] - m); sum += e[k]; }
            float isum = 1.0f / sum;
            for (int k = 0; k < LB; ++k) ps[t * LB + k] = e[k] * isum;
        }
        __syncthreads();
        for (int e = t; e < LB * DHB; e += 256) {
            int qi = e / DHB, d = e % DHB;
            float s = 0.0f;
            for (int k = 0; k < LB; ++k) s += ps[qi * LB + k] * vs[k * DHB + d];
            ctx16[base + (size_t)qi * HB + h0 + d] = f2b(s);
        }
        __syncthreads();
    }
}

// ---------------------------------------------------------------------------
// Layer-2 flash attention: block = (b, h, qblock of 128). 8 waves, each wave
// owns 16 q rows. K/V streamed in 32-col tiles through LDS (V via async
// global->LDS DMA, K transposed on store). Per iteration: 6 WMMAs for
// S = Q*K^T, online softmax, P->LDS->A-fragment, 6 WMMAs for O += P*V.
// ---------------------------------------------------------------------------
#define KTP 40    // 32+8
#define VTP 104   // 96+8

__global__ __launch_bounds__(256) void flash_attn_kernel(const unsigned short* __restrict__ Q,
                                                         const unsigned short* __restrict__ K,
                                                         const unsigned short* __restrict__ V,
                                                         const float* __restrict__ am,
                                                         unsigned short* __restrict__ ctx16) {
    __shared__ unsigned short ktLds[DHB * KTP];   // K^T [96][32+pad]
    __shared__ unsigned short vtLds[32 * VTP];    // V   [32][96+pad]
    __shared__ unsigned short pLds[8][16 * KTP];  // per-wave P 16x32
    int b = blockIdx.x, h = blockIdx.y, qb = blockIdx.z;
    int h0 = h * DHB;
    int t = threadIdx.x;
    int w = t >> 5, l = t & 31, r = l & 15, hi = l >> 4;
    int q0 = qb * 128 + w * 16;
    const float scale = 0.1020620726159658f;

    // Q fragments (A layout) straight from global
    BF16Frag qf[3];
    {
        size_t qrow = ((size_t)b * SB + q0 + r) * HB + h0;
        for (int c = 0; c < 3; ++c) {
            const unsigned short* p = Q + qrow + c * 32 + hi * 8;
            qf[c].u4[0] = *(const uint4*)p;
            qf[c].u4[1] = *(const uint4*)(p + 16);
        }
    }

    const v8f vzero = {0.f, 0.f, 0.f, 0.f, 0.f, 0.f, 0.f, 0.f};
    v8f cacc[6];
    for (int f = 0; f < 6; ++f) cacc[f] = vzero;
    float mOld[8], lSum[8];
    for (int rr = 0; rr < 8; ++rr) { mOld[rr] = -INFINITY; lSum[rr] = 0.0f; }

    for (int kt0 = 0; kt0 < SB; kt0 += 32) {
        // V tile: async DMA global->LDS, 16B chunks (rows are contiguous)
        for (int e = t; e < 32 * 12; e += 256) {
            int j = e / 12, cc = e % 12;
            const unsigned short* g = V + ((size_t)b * SB + kt0 + j) * HB + h0 + cc * 8;
            unsigned int lofs = (unsigned int)(size_t)&vtLds[j * VTP + cc * 8];
            asm volatile("global_load_async_to_lds_b128 %0, %1, off"
                         :: "v"(lofs), "v"(g) : "memory");
        }
        // K tile: transpose on store (feature-major for B fragments)
        for (int e = t; e < 32 * DHB; e += 256) {
            int j = e / DHB, d = e % DHB;
            ktLds[d * KTP + j] = K[((size_t)b * SB + kt0 + j) * HB + h0 + d];
        }
        asm volatile("s_wait_asynccnt 0x0" ::: "memory");
        __syncthreads();

        // S = Q (16x96) * K^T  -> two 16x16 accs (k columns 0..15, 16..31)
        v8f sacc[2] = {vzero, vzero};
        for (int c = 0; c < 3; ++c) {
            BF16Frag bk0, bk1;
            const unsigned short* p = &ktLds[(c * 32 + hi * 8) * KTP + r];
            for (int s = 0; s < 8; ++s) {
                bk0.s[s] = p[s * KTP];             bk0.s[8 + s] = p[(16 + s) * KTP];
                bk1.s[s] = p[s * KTP + 16];        bk1.s[8 + s] = p[(16 + s) * KTP + 16];
            }
            sacc[0] = wmma_bf16(qf[c].v, bk0.v, sacc[0]);
            sacc[1] = wmma_bf16(qf[c].v, bk1.v, sacc[1]);
        }

        float mCol0 = (1.0f - am[b * SB + kt0 + r]) * -10000.0f;
        float mCol1 = (1.0f - am[b * SB + kt0 + 16 + r]) * -10000.0f;

        unsigned short* pw = &pLds[w][0];
        for (int rr = 0; rr < 8; ++rr) {
            float s0 = sacc[0][rr] * scale + mCol0;
            float s1 = sacc[1][rr] * scale + mCol1;
            float m = fmaxf(s0, s1);
            for (int off = 1; off < 16; off <<= 1) m = fmaxf(m, __shfl_xor(m, off, 32));
            float mNew = fmaxf(mOld[rr], m);
            float e0 = expf(s0 - mNew), e1 = expf(s1 - mNew);
            float ts = e0 + e1;
            for (int off = 1; off < 16; off <<= 1) ts += __shfl_xor(ts, off, 32);
            float f = expf(mOld[rr] - mNew);
            lSum[rr] = lSum[rr] * f + ts;
            mOld[rr] = mNew;
            for (int fa = 0; fa < 6; ++fa) cacc[fa][rr] *= f;
            int mrow = rr + 8 * hi;                       // C/D row mapping
            pw[mrow * KTP + r] = f2b(e0);                 // cols 0..15
            pw[mrow * KTP + 16 + r] = f2b(e1);            // cols 16..31
        }

        // P (16x32) as A fragment
        BF16Frag pa;
        const unsigned short* pp = pw + r * KTP + hi * 8;
        pa.u4[0] = *(const uint4*)pp;
        pa.u4[1] = *(const uint4*)(pp + 16);

        // O += P * V  (6 WMMAs over feature dim 96)
        for (int fa = 0; fa < 6; ++fa) {
            BF16Frag bv;
            const unsigned short* p = &vtLds[hi * 8 * VTP + fa * 16 + r];
            for (int s = 0; s < 8; ++s) {
                bv.s[s] = p[s * VTP];
                bv.s[8 + s] = p[(16 + s) * VTP];
            }
            cacc[fa] = wmma_bf16(pa.v, bv.v, cacc[fa]);
        }
        __syncthreads();
    }

    for (int rr = 0; rr < 8; ++rr) {
        float inv = 1.0f / lSum[rr];
        size_t orow = ((size_t)b * SB + q0 + rr + 8 * hi) * HB + h0;
        for (int fa = 0; fa < 6; ++fa)
            ctx16[orow + fa * 16 + r] = f2b(cacc[fa][rr] * inv);
    }
}

// ---------------------------------------------------------------------------
// Host orchestration
// ---------------------------------------------------------------------------
extern "C" void kernel_launch(void* const* d_in, const int* in_sizes, int n_in,
                              void* d_out, int out_size, void* d_ws, size_t ws_size,
                              hipStream_t stream) {
    const float* tlf = (const float*)d_in[0];
    const int*   cl  = (const int*)d_in[1];
    const float* am  = (const float*)d_in[2];
    const float* Wq  = (const float*)d_in[3];
    const float* bq  = (const float*)d_in[4];
    const float* Wk  = (const float*)d_in[5];
    const float* bk  = (const float*)d_in[6];
    const float* Wv  = (const float*)d_in[7];
    const float* bv  = (const float*)d_in[8];
    const float* Wo  = (const float*)d_in[9];
    const float* bo  = (const float*)d_in[10];
    const float* g1  = (const float*)d_in[11];
    const float* b1  = (const float*)d_in[12];
    const float* Wi  = (const float*)d_in[13];
    const float* bi  = (const float*)d_in[14];
    const float* Wd  = (const float*)d_in[15];
    const float* bd  = (const float*)d_in[16];
    const float* g2  = (const float*)d_in[17];
    const float* b2  = (const float*)d_in[18];
    float* out = (float*)d_out;

    char* ws = (char*)d_ws;
    size_t off = 0;
    auto alloc = [&](size_t bytes) -> void* {
        void* p = ws + off;
        off = (off + bytes + 255) & ~(size_t)255;
        return p;
    };
    const size_t HH = (size_t)HB * HB;          // 589824
    const size_t HF = (size_t)HB * FFB;         // 2359296
    const size_t RH = (size_t)MROWS * HB;       // 6291456
    const size_t RF = (size_t)MROWS * FFB;      // 25165824

    unsigned short* Wq16 = (unsigned short*)alloc(2 * HH * 2);
    unsigned short* Wk16 = (unsigned short*)alloc(2 * HH * 2);
    unsigned short* Wv16 = (unsigned short*)alloc(2 * HH * 2);
    unsigned short* Wo16 = (unsigned short*)alloc(2 * HH * 2);
    unsigned short* Wi16 = (unsigned short*)alloc(2 * HF * 2);
    unsigned short* Wd16 = (unsigned short*)alloc(2 * HF * 2);
    float*          XG   = (float*)alloc(RH * 4);
    unsigned short* XG16 = (unsigned short*)alloc(RH * 2);
    unsigned short* Q16  = (unsigned short*)alloc(RH * 2);
    unsigned short* K16  = (unsigned short*)alloc(RH * 2);
    unsigned short* V16  = (unsigned short*)alloc(RH * 2);
    unsigned short* C16  = (unsigned short*)alloc(RH * 2);
    float*          AP   = (float*)alloc(RH * 4);
    float*          LN1  = (float*)alloc(RH * 4);
    unsigned short* LN116 = (unsigned short*)alloc(RH * 2);
    unsigned short* INT16 = (unsigned short*)alloc(RF * 2);
    float*          FFN  = (float*)alloc(RH * 4);

    auto cvt = [&](const float* s, unsigned short* d, long n) {
        int g = (int)((n + 255) / 256);
        if (g > 4096) g = 4096;
        cvt_kernel<<<g, 256, 0, stream>>>(s, d, n);
    };
    cvt(Wq, Wq16, 2 * HH);
    cvt(Wk, Wk16, 2 * HH);
    cvt(Wv, Wv16, 2 * HH);
    cvt(Wo, Wo16, 2 * HH);
    cvt(Wi, Wi16, 2 * HF);
    cvt(Wd, Wd16, 2 * HF);

    auto gemm = [&](const unsigned short* A, const unsigned short* B, const float* bias,
                    float* oF, unsigned short* oB, int M, int N, int Kd, int gelu) {
        dim3 g(M / 128, N / 128);
        gemm_bf16_kernel<<<g, 256, 0, stream>>>(A, B, bias, oF, oB, M, N, Kd, gelu);
    };

    gather_kernel<<<MB * CB, 256, 0, stream>>>(tlf, cl, XG, XG16);

    for (int li = 0; li < 2; ++li) {
        const unsigned short* wq = Wq16 + (size_t)li * HH;
        const unsigned short* wk = Wk16 + (size_t)li * HH;
        const unsigned short* wv = Wv16 + (size_t)li * HH;
        const unsigned short* wo = Wo16 + (size_t)li * HH;
        const unsigned short* wi = Wi16 + (size_t)li * HF;
        const unsigned short* wd = Wd16 + (size_t)li * HF;

        gemm(XG16, wq, bq + li * HB, nullptr, Q16, MROWS, HB, HB, 0);
        gemm(XG16, wk, bk + li * HB, nullptr, K16, MROWS, HB, HB, 0);
        gemm(XG16, wv, bv + li * HB, nullptr, V16, MROWS, HB, HB, 0);

        if (li == 0) {
            chunk_attn_kernel<<<MB * CB, 256, 0, stream>>>(Q16, K16, V16, cl, C16);
        } else {
            flash_attn_kernel<<<dim3(MB, NHB, 4), 256, 0, stream>>>(Q16, K16, V16, am, C16);
        }

        gemm(C16, wo, bo + li * HB, AP, nullptr, MROWS, HB, HB, 0);
        ln_kernel<<<MROWS, 256, 0, stream>>>(AP, XG, g1 + li * HB, b1 + li * HB, LN1, LN116);
        gemm(LN116, wi, bi + li * FFB, nullptr, INT16, MROWS, FFB, HB, 1);   // GELU
        gemm(INT16, wd, bd + li * HB, FFN, nullptr, MROWS, HB, FFB, 0);

        if (li == 0) {
            ln_kernel<<<MROWS, 256, 0, stream>>>(FFN, LN1, g2 + li * HB, b2 + li * HB, AP, nullptr);
            hipMemsetAsync(XG, 0, RH * 4, stream);
            hipMemsetAsync(XG16, 0, RH * 2, stream);
            scatter_kernel<<<MB * CB, 256, 0, stream>>>(AP, cl, XG, XG16);
        } else {
            ln_kernel<<<MROWS, 256, 0, stream>>>(FFN, LN1, g2 + li * HB, b2 + li * HB, out, nullptr);
        }
    }
}
